// HCA_28475633172693
// MI455X (gfx1250) — matrix operand order repaired and checked by reading, not compiled, not run on hardware
//
#include <hip/hip_runtime.h>

// ---------------------------------------------------------------------------
// Types / helpers
// ---------------------------------------------------------------------------
typedef __attribute__((ext_vector_type(16))) __bf16 v16bf;
typedef __attribute__((ext_vector_type(8)))  float  v8f;

struct alignas(16) U4 { unsigned int x, y, z, w; };

union Frag16 { v16bf v; U4 u[2]; unsigned short s[16]; };

__device__ __forceinline__ unsigned short f2bf(float f) {
  union { float f; unsigned int u; } c; c.f = f;
  unsigned int u = c.u;
  u += 0x7FFFu + ((u >> 16) & 1u);   // round-to-nearest-even
  return (unsigned short)(u >> 16);
}

__device__ __forceinline__ v8f wmma_bf16(v16bf a, v16bf b, v8f acc) {
  // D(f32,16x16) = A(bf16,16x32) * B(bf16,32x16) + C
  return __builtin_amdgcn_wmma_f32_16x16x32_bf16(false, a, false, b,
                                                 (short)0, acc, false, false);
}

// B fragment (K-slice of 32 at 'koff') from a row-major B^T [N][K] matrix:
// lane n = column, halves j -> K = koff + (lane>>4)*16 + j  (contiguous).
__device__ __forceinline__ v16bf load_bfragT(const unsigned short* btrow,
                                             int koff, int ks16) {
  Frag16 f;
  const unsigned short* p = btrow + koff + ks16;
  f.u[0] = *(const U4*)(p);
  f.u[1] = *(const U4*)(p + 8);
  return f.v;
}

// CDNA5 async DMA: global -> LDS, 16 bytes per lane, tracked by ASYNCcnt.
__device__ __forceinline__ void async_load_lds_b128(unsigned lds_byte_off,
                                                    const void* gaddr) {
  asm volatile("global_load_async_to_lds_b128 %0, %1, off"
               :: "v"(lds_byte_off), "v"(gaddr) : "memory");
}
__device__ __forceinline__ void wait_async0() {
  asm volatile("s_wait_asynccnt 0x0" ::: "memory");
}

// ---------------------------------------------------------------------------
// f32 -> bf16 cast (grid-stride)
// ---------------------------------------------------------------------------
__global__ void cast_f32_bf16_kernel(const float* __restrict__ in,
                                     unsigned short* __restrict__ out,
                                     long long n) {
  long long i = (long long)blockIdx.x * blockDim.x + threadIdx.x;
  long long stride = (long long)gridDim.x * blockDim.x;
  for (; i < n; i += stride) out[i] = f2bf(in[i]);
}

// ---------------------------------------------------------------------------
// f32 [K][N] -> bf16 transposed [N][K]   (32x32 tiles through LDS)
// grid = (N/32, K/32), 256 threads.
// ---------------------------------------------------------------------------
__global__ __launch_bounds__(256) void cast_transpose_kernel(
    const float* __restrict__ in, unsigned short* __restrict__ out,
    int K, int N) {
  __shared__ unsigned short tile[32][33];   // [n_local][k_local]
  const int kt = blockIdx.y * 32, nt = blockIdx.x * 32;
  const int r = threadIdx.x >> 3;           // 0..31
  const int c4 = (threadIdx.x & 7) * 4;     // 0..28
#pragma unroll
  for (int i = 0; i < 4; ++i)
    tile[c4 + i][r] = f2bf(in[(size_t)(kt + r) * N + nt + c4 + i]);
  __syncthreads();
#pragma unroll
  for (int i = 0; i < 4; ++i)
    out[(size_t)(nt + r) * K + kt + c4 + i] = tile[r][c4 + i];
}

// ---------------------------------------------------------------------------
// bf16 GEMM: C = A[MxK] (row-major) * B (given as B^T [N][K] row-major).
// Templated on MF (M-fragments per wave): block tile (MF*64) x 128,
// 256 threads = 8 waves (4x2), wave tile (MF*16*... rows = MF*16)x64:
//   MF=2: 128x128 block, 8 WMMA/k-step/wave   (small-M GEMMs)
//   MF=4: 256x128 block, 16 WMMA/k-step/wave  (big GEMMs: 2x FLOPs per byte)
// A tile double-buffered in LDS via GLOBAL_LOAD_ASYNC_TO_LDS_B128 (ASYNCcnt),
// next tile's DMA issued unconditionally (wrapped) to keep control flow flat.
// B frags read directly from global B^T (contiguous b128, L2-resident weights).
// Outputs: Cf (f32), Cb (bf16), Cbt (bf16 transposed [N][M], ldct) - nullable.
// Requires M % (MF*64) == 0, N%128==0, K%32==0, lda/ldc multiples of 8.
// ---------------------------------------------------------------------------
#define GA_STRIDE 40    // halves; 80B rows -> conflict-free b128 A frag reads

template <int MF>
__global__ __launch_bounds__(256) void gemm_bf16_kernel(
    const unsigned short* __restrict__ A, int lda,
    const unsigned short* __restrict__ BT, int ldbt,
    float* __restrict__ Cf, unsigned short* __restrict__ Cb,
    unsigned short* __restrict__ Cbt, int ldc, int ldct, int K) {
  constexpr int ROWS = MF * 64;                    // A-tile rows per block
  constexpr int BUFB = ROWS * GA_STRIDE * 2;       // bytes per LDS buffer
  __shared__ __align__(16) unsigned short sA[2 * ROWS * GA_STRIDE];

  const int tid  = threadIdx.x;
  const int lane = tid & 31;
  const int w    = tid >> 5;
  const int wrow = w >> 1;              // 0..3
  const int wcol = w & 1;               // 0..1
  const int bm0  = blockIdx.y * ROWS;
  const int bn0  = blockIdx.x * 128;

  const v8f zero8 = {0.f,0.f,0.f,0.f,0.f,0.f,0.f,0.f};
  v8f acc[MF][4];
#pragma unroll
  for (int i = 0; i < MF; ++i)
#pragma unroll
    for (int j = 0; j < 4; ++j) acc[i][j] = zero8;

  // A staging: MF==2 -> 2 threads/row (2xb128 each); MF==4 -> 1 thread/row (4xb128)
  constexpr int CH = (MF == 2) ? 2 : 4;            // 16B chunks per thread
  const int ar = (MF == 2) ? (tid >> 1) : tid;
  const int ac = (MF == 2) ? ((tid & 1) * 16) : 0;

  const int m_lane = lane & 15;
  const int ksel8  = (lane >> 4) * 8;
  const int ksel16 = (lane >> 4) * 16;

  const unsigned short* Arow = A + (size_t)(bm0 + ar) * lda + ac;
  const unsigned stage_off =
      (unsigned)(unsigned long long)(sA + ar * GA_STRIDE + ac);

  // per-nf B^T row pointers (loop-invariant)
  const unsigned short* btrow[4];
#pragma unroll
  for (int nf = 0; nf < 4; ++nf)
    btrow[nf] = BT + (size_t)(bn0 + wcol * 64 + nf * 16 + m_lane) * ldbt;

  // prologue: DMA tile 0 into buffer 0
#pragma unroll
  for (int c = 0; c < CH; ++c)
    async_load_lds_b128(stage_off + 16 * c, Arow + 8 * c);

  int parity = 0;
  for (int kt = 0; kt < K; kt += 32, parity ^= 1) {
    wait_async0();          // own DMA for this tile complete
    __syncthreads();        // everyone's DMA visible
    {                       // unconditional next-tile DMA (wrapped on last iter)
      int ktn = kt + 32; if (ktn >= K) ktn = 0;
      unsigned soff = stage_off + (parity ^ 1) * BUFB;
      const unsigned short* gnext = Arow + ktn;
#pragma unroll
      for (int c = 0; c < CH; ++c)
        async_load_lds_b128(soff + 16 * c, gnext + 8 * c);
    }

    const unsigned short* sAc = sA + parity * (ROWS * GA_STRIDE);
    v16bf afrag[MF];
#pragma unroll
    for (int mf = 0; mf < MF; ++mf) {
      const unsigned short* p =
          sAc + (wrow * (MF * 16) + mf * 16 + m_lane) * GA_STRIDE;
      Frag16 f;
      f.u[0] = *(const U4*)(p + ksel16);
      f.u[1] = *(const U4*)(p + ksel16 + 8);
      afrag[mf] = f.v;
    }
#pragma unroll
    for (int nf = 0; nf < 4; ++nf) {
      v16bf bfrag = load_bfragT(btrow[nf], kt, ksel16);
#pragma unroll
      for (int mf = 0; mf < MF; ++mf)
        acc[mf][nf] = wmma_bf16(afrag[mf], bfrag, acc[mf][nf]);
    }
  }

  const int half8 = (lane >> 4) * 8;
#pragma unroll
  for (int mf = 0; mf < MF; ++mf)
#pragma unroll
    for (int nf = 0; nf < 4; ++nf)
#pragma unroll
      for (int v = 0; v < 8; ++v) {
        int gm = bm0 + wrow * (MF * 16) + mf * 16 + half8 + v;
        int gn = bn0 + wcol * 64 + nf * 16 + m_lane;
        float val = acc[mf][nf][v];
        if (Cf)  Cf[(size_t)gm * ldc + gn] = val;
        if (Cb)  Cb[(size_t)gm * ldc + gn] = f2bf(val);
        if (Cbt) Cbt[(size_t)gn * ldct + gm] = f2bf(val);
      }
}

// ---------------------------------------------------------------------------
// Partial RoPE (first 64 dims, halves 32/32) + RMS norm over 128 dims,
// f32 in -> bf16 out (+ optional transposed bf16 out).
// grid = (rows, heads), block = 128.
// in index: t*in_ld + h*128 + d ; out: h*out_sh + t*out_st + d ; out_t: d*ldt + t
// ---------------------------------------------------------------------------
__global__ void rope_rms_kernel(const float* __restrict__ in, int in_ld,
                                const float* __restrict__ w,
                                unsigned short* __restrict__ out,
                                int out_st, int out_sh, int do_rope,
                                unsigned short* __restrict__ out_t, int ldt) {
  const int t = blockIdx.x, h = blockIdx.y, d = threadIdx.x;
  __shared__ float sv[128];
  float val = in[(size_t)t * in_ld + h * 128 + d];
  sv[d] = val;
  __syncthreads();
  float r = val;
  if (do_rope && d < 64) {
    int i = d & 31;
    float inv = expf(-(float)i * (9.210340371976184f / 32.0f)); // 10000^(-i/32)
    float ang = (float)t * inv;
    float s, c; sincosf(ang, &s, &c);
    float x1 = sv[i], x2 = sv[i + 32];
    r = (d < 32) ? (x1 * c - x2 * s) : (x2 * c + x1 * s);
  }
  __syncthreads();
  sv[d] = r;
  __syncthreads();
  float ss = 0.f;
#pragma unroll 8
  for (int j = 0; j < 128; ++j) { float x = sv[j]; ss += x * x; }
  float rs = rsqrtf(ss * (1.0f / 128.0f) + 1e-6f);
  unsigned short o = f2bf(r * rs * w[d]);
  out[(size_t)h * out_sh + (size_t)t * out_st + d] = o;
  if (out_t) out_t[(size_t)d * ldt + t] = o;
}

// ---------------------------------------------------------------------------
// Flash attention with sink, WMMA bf16, no LDS staging (direct global frags).
// Q: [NH][T][128] bf16 ; Km: [S][128] bf16 ; VT: [128][S] bf16 (transposed V);
// X: [T][NH*128] f32.
// mode 0: compressed (key j valid iff j*16+15 < i); mode 1: sliding (j<=i, i-j<512).
// grid = (T/128, NH), 256 threads = 8 waves, one 16-query tile per wave.
// Sink folded into softmax init: m = sink[h], l = 1.
// ---------------------------------------------------------------------------
#define SWIN 512
#define P_STRIDE 40   // halves; 80B rows: b128-aligned & conflict-free

__global__ __launch_bounds__(256) void attn_kernel(
    const unsigned short* __restrict__ Q,
    const unsigned short* __restrict__ Km,
    const unsigned short* __restrict__ VT,
    const float* __restrict__ sink,
    float* __restrict__ X,
    int S, int mode, int accumulate) {
  __shared__ __align__(16) unsigned short sP[8 * 16 * P_STRIDE];

  const int tid = threadIdx.x, lane = tid & 31, w = tid >> 5;
  const int h = blockIdx.y;
  const int qb = blockIdx.x * 128 + w * 16;
  const float scale = 0.08838834764831845f; // 128^-0.5
  unsigned short* sPw = sP + w * 16 * P_STRIDE;

  const int nl = lane & 15;
  const int half8 = (lane >> 4) * 8;
  const int ks8 = (lane >> 4) * 8;
  const int ks16 = (lane >> 4) * 16;

  // preload the 4 Q A-fragments (HD=128 -> 4 K-chunks of 32)
  v16bf aq[4];
  {
    const unsigned short* p = Q + ((size_t)h * 2048 + qb + nl) * 128;
#pragma unroll
    for (int kc = 0; kc < 4; ++kc) {
      Frag16 f;
      f.u[0] = *(const U4*)(p + kc * 32 + ks16);
      f.u[1] = *(const U4*)(p + kc * 32 + ks16 + 8);
      aq[kc] = f.v;
    }
  }

  const v8f zero8 = {0.f,0.f,0.f,0.f,0.f,0.f,0.f,0.f};
  float snk = sink[h];
  float mrow[8], lrow[8];
  v8f O[8];
#pragma unroll
  for (int v = 0; v < 8; ++v) { mrow[v] = snk; lrow[v] = 1.0f; }
#pragma unroll
  for (int c = 0; c < 8; ++c) O[c] = zero8;

  // per-wave key range (no cross-wave sharing -> no barriers)
  int kb_start = 0, kb_end = S - 1;
  if (mode == 1) {
    int lo = qb - (SWIN - 1); if (lo < 0) lo = 0;
    kb_start = lo & ~31;
    kb_end = qb + 15;
  }

  const unsigned short* krow0 = Km + (size_t)nl * 128;        // keys kb+nl
  const unsigned short* krow1 = Km + (size_t)(16 + nl) * 128; // keys kb+16+nl

  for (int kb = kb_start; kb <= kb_end; kb += 32) {
    // ---- S = Q * K^T : two 16-key tiles; contiguous b128 reads of K rows ----
    v8f s0 = zero8, s1 = zero8;
#pragma unroll
    for (int kc = 0; kc < 4; ++kc) {
      Frag16 f0, f1;
      const unsigned short* p0 = krow0 + (size_t)kb * 128 + kc * 32 + ks16;
      const unsigned short* p1 = krow1 + (size_t)kb * 128 + kc * 32 + ks16;
      f0.u[0] = *(const U4*)(p0); f0.u[1] = *(const U4*)(p0 + 8);
      f1.u[0] = *(const U4*)(p1); f1.u[1] = *(const U4*)(p1 + 8);
      s0 = wmma_bf16(aq[kc], f0.v, s0);
      s1 = wmma_bf16(aq[kc], f1.v, s1);
    }

    // ---- online softmax (row stats across 16 lanes of each half-wave) ----
    float p0v[8], p1v[8], alpha[8];
#pragma unroll
    for (int v = 0; v < 8; ++v) {
      int ig = qb + half8 + v;
      float x0 = s0[v] * scale, x1 = s1[v] * scale;
      int k0i = kb + nl, k1i = kb + 16 + nl;
      bool ok0, ok1;
      if (mode == 0) { ok0 = (k0i * 16 + 15) < ig; ok1 = (k1i * 16 + 15) < ig; }
      else { ok0 = (k0i <= ig) && (ig - k0i < SWIN);
             ok1 = (k1i <= ig) && (ig - k1i < SWIN); }
      x0 = ok0 ? x0 : -1e30f;
      x1 = ok1 ? x1 : -1e30f;
      float mx = fmaxf(x0, x1);
      mx = fmaxf(mx, __shfl_xor(mx, 1));
      mx = fmaxf(mx, __shfl_xor(mx, 2));
      mx = fmaxf(mx, __shfl_xor(mx, 4));
      mx = fmaxf(mx, __shfl_xor(mx, 8));
      float mnew = fmaxf(mrow[v], mx);
      float al = __expf(mrow[v] - mnew);
      float e0 = __expf(x0 - mnew), e1 = __expf(x1 - mnew);
      float rs = e0 + e1;
      rs += __shfl_xor(rs, 1);
      rs += __shfl_xor(rs, 2);
      rs += __shfl_xor(rs, 4);
      rs += __shfl_xor(rs, 8);
      lrow[v] = lrow[v] * al + rs;
      mrow[v] = mnew; alpha[v] = al; p0v[v] = e0; p1v[v] = e1;
    }
#pragma unroll
    for (int c = 0; c < 8; ++c)
#pragma unroll
      for (int v = 0; v < 8; ++v) O[c][v] = O[c][v] * alpha[v];

    // ---- P (C-layout) -> A-layout via per-wave LDS scratch ----
#pragma unroll
    for (int v = 0; v < 8; ++v) {
      sPw[(half8 + v) * P_STRIDE + nl]      = f2bf(p0v[v]);
      sPw[(half8 + v) * P_STRIDE + 16 + nl] = f2bf(p1v[v]);
    }
    asm volatile("s_wait_dscnt 0" ::: "memory");
    v16bf pA;
    {
      Frag16 f;
      const unsigned short* pr = sPw + nl * P_STRIDE;
      f.u[0] = *(const U4*)(pr + ks16);
      f.u[1] = *(const U4*)(pr + ks16 + 8);
      pA = f.v;
    }

    // ---- O += P(16x32) * V(32x128) : contiguous b128 reads of V^T rows ----
#pragma unroll
    for (int c = 0; c < 8; ++c) {
      Frag16 f;
      const unsigned short* pv = VT + (size_t)(c * 16 + nl) * S + kb + ks16;
      f.u[0] = *(const U4*)(pv); f.u[1] = *(const U4*)(pv + 8);
      O[c] = wmma_bf16(pA, f.v, O[c]);
    }
  }

  // ---- epilogue: X[t][h*128 + d] (+)= O / l ----
#pragma unroll
  for (int c = 0; c < 8; ++c)
#pragma unroll
    for (int v = 0; v < 8; ++v) {
      int t = qb + half8 + v;
      int col = h * 128 + c * 16 + nl;
      float val = O[c][v] / lrow[v];
      size_t idx = (size_t)t * 2048 + col;
      if (accumulate) X[idx] += val; else X[idx] = val;
    }
}

// ---------------------------------------------------------------------------
// Host launcher
// ---------------------------------------------------------------------------
extern "C" void kernel_launch(void* const* d_in, const int* in_sizes, int n_in,
                              void* d_out, int out_size, void* d_ws, size_t ws_size,
                              hipStream_t stream) {
  (void)in_sizes; (void)n_in; (void)out_size; (void)ws_size;
  const float* h    = (const float*)d_in[0];
  const float* Wqc  = (const float*)d_in[1];
  const float* Wqu  = (const float*)d_in[2];
  const float* Wc   = (const float*)d_in[3];
  const float* Wk   = (const float*)d_in[4];
  const float* Wv   = (const float*)d_in[5];
  const float* qn_w = (const float*)d_in[6];
  const float* kn_w = (const float*)d_in[7];
  const float* sink = (const float*)d_in[8];
  const float* Wg   = (const float*)d_in[9];
  const float* Wo   = (const float*)d_in[10];
  float* out = (float*)d_out;

  const int T = 2048, HID = 2048, NH = 16, HD = 128, QC = 512, NB = 128;
  const int G = 4, AI = 2048;
  const int KC = 16 * HID;            // 32768

  char* ws = (char*)d_ws;
  size_t off = 0;
  auto alloc = [&](size_t bytes) -> void* {
    void* p = ws + off;
    off += (bytes + 255) & ~(size_t)255;
    return p;
  };

  unsigned short* h_bf     = (unsigned short*)alloc((size_t)T * HID * 2);
  unsigned short* WqcT     = (unsigned short*)alloc((size_t)HID * QC * 2);   // [QC][HID]
  unsigned short* WquT     = (unsigned short*)alloc((size_t)QC * HID * 2);   // [HID][QC]
  unsigned short* WcT      = (unsigned short*)alloc((size_t)KC * HD * 2);    // [HD][KC]
  unsigned short* WkT      = (unsigned short*)alloc((size_t)HID * HD * 2);   // [HD][HID]
  unsigned short* WvT      = (unsigned short*)alloc((size_t)HID * HD * 2);   // [HD][HID]
  unsigned short* WgT      = (unsigned short*)alloc((size_t)G * 512 * AI * 2); // per g: [AI][512]
  unsigned short* WoT      = (unsigned short*)alloc((size_t)G * AI * HID * 2); // [HID][G*AI]
  unsigned short* qpre_bf  = (unsigned short*)alloc((size_t)T * QC * 2);
  float*          q_f32    = (float*)alloc((size_t)T * HID * 4);
  unsigned short* q_bf     = (unsigned short*)alloc((size_t)NH * T * HD * 2);
  float*          ckv_f32  = (float*)alloc((size_t)NB * HD * 4);
  unsigned short* ckv_bf   = (unsigned short*)alloc((size_t)NB * HD * 2);
  unsigned short* ckv_bfT  = (unsigned short*)alloc((size_t)NB * HD * 2);
  float*          k_f32    = (float*)alloc((size_t)T * HD * 4);
  unsigned short* k_bf     = (unsigned short*)alloc((size_t)T * HD * 2);
  unsigned short* v_bfT    = (unsigned short*)alloc((size_t)T * HD * 2);     // [HD][T]
  float*          x_f32    = (float*)alloc((size_t)T * HID * 4);
  unsigned short* x_bf     = (unsigned short*)alloc((size_t)T * HID * 2);
  unsigned short* inter_bf = (unsigned short*)alloc((size_t)T * G * AI * 2);

  auto cast = [&](const float* src, unsigned short* dst, long long n) {
    int blocks = (int)((n + 256LL * 16 - 1) / (256LL * 16));
    if (blocks > 4096) blocks = 4096;
    if (blocks < 1) blocks = 1;
    cast_f32_bf16_kernel<<<blocks, 256, 0, stream>>>(src, dst, n);
  };
  auto castT = [&](const float* src, unsigned short* dstT, int K, int N) {
    dim3 grid(N / 32, K / 32);
    cast_transpose_kernel<<<grid, 256, 0, stream>>>(src, dstT, K, N);
  };
  auto gemm = [&](const unsigned short* A, int lda, const unsigned short* BT,
                  int ldbt, float* Cf, unsigned short* Cb, unsigned short* Cbt,
                  int ldc, int ldct, int M, int N, int K) {
    if (M % 256 == 0) {
      dim3 grid(N / 128, M / 256);
      gemm_bf16_kernel<4><<<grid, 256, 0, stream>>>(A, lda, BT, ldbt, Cf, Cb,
                                                    Cbt, ldc, ldct, K);
    } else {
      dim3 grid(N / 128, M / 128);
      gemm_bf16_kernel<2><<<grid, 256, 0, stream>>>(A, lda, BT, ldbt, Cf, Cb,
                                                    Cbt, ldc, ldct, K);
    }
  };

  // --- casts: activations plain, weights transposed ---
  cast(h, h_bf, (long long)T * HID);
  castT(Wqc, WqcT, HID, QC);
  castT(Wqu, WquT, QC, HID);
  castT(Wc,  WcT,  KC,  HD);
  castT(Wk,  WkT,  HID, HD);
  castT(Wv,  WvT,  HID, HD);
  for (int g = 0; g < G; ++g)
    castT(Wg + (size_t)g * 512 * AI, WgT + (size_t)g * 512 * AI, 512, AI);
  castT(Wo, WoT, G * AI, HID);

  // --- Q path: h@Wqc -> @Wqu -> rope+rms -> q_bf [NH][T][HD] ---
  gemm(h_bf, HID, WqcT, HID, nullptr, qpre_bf, nullptr, QC, 0, T, QC, HID);
  gemm(qpre_bf, QC, WquT, QC, q_f32, nullptr, nullptr, HID, 0, T, HID, QC);
  rope_rms_kernel<<<dim3(T, NH), 128, 0, stream>>>(q_f32, HID, qn_w, q_bf,
                                                   HD, T * HD, 1, nullptr, 0);

  // --- compressed KV: (h viewed [128][32768]) @ Wc -> rms -> ckv(+T) ---
  gemm(h_bf, KC, WcT, KC, ckv_f32, nullptr, nullptr, HD, 0, NB, HD, KC);
  rope_rms_kernel<<<dim3(NB, 1), 128, 0, stream>>>(ckv_f32, HD, kn_w, ckv_bf,
                                                   HD, 0, 0, ckv_bfT, NB);

  // --- sliding-window K/V ---
  gemm(h_bf, HID, WkT, HID, k_f32, nullptr, nullptr, HD, 0, T, HD, HID);
  rope_rms_kernel<<<dim3(T, 1), 128, 0, stream>>>(k_f32, HD, kn_w, k_bf,
                                                  HD, 0, 1, nullptr, 0);
  gemm(h_bf, HID, WvT, HID, nullptr, nullptr, v_bfT, HD, T, T, HD, HID);

  // --- attentions (sink folded into softmax init) ---
  attn_kernel<<<dim3(T / 128, NH), 256, 0, stream>>>(q_bf, ckv_bf, ckv_bfT,
                                                     sink, x_f32, NB, 0, 0);
  attn_kernel<<<dim3(T / 128, NH), 256, 0, stream>>>(q_bf, k_bf, v_bfT,
                                                     sink, x_f32, T, 1, 1);

  // --- output projections ---
  cast(x_f32, x_bf, (long long)T * HID);
  for (int g = 0; g < G; ++g)
    gemm(x_bf + g * 512, HID, WgT + (size_t)g * 512 * AI, 512,
         nullptr, inter_bf + g * AI, nullptr, G * AI, 0, T, AI, 512);
  gemm(inter_bf, G * AI, WoT, G * AI, out, nullptr, nullptr, HID, 0,
       T, HID, G * AI);
}
